// ImportanceRenderer_1211180777618
// MI455X (gfx1250) — matrix operand order repaired and checked by reading, not compiled, not run on hardware
//
#include <hip/hip_runtime.h>

#define NBATCH 4
#define NRAYS  4096
#define TOTAL_RAYS (NBATCH*NRAYS)
#define CCH   32
#define HRES  256
#define SAMP  48
#define NSEG  47
#define RAYS_PER_WG 8
#define WAVES 4
#define BLOCK 128
#define TILES_PER_WAVE ((RAYS_PER_WG*SAMP)/(16*WAVES))   /* = 6 */

#define RAY_T0 2.25f
#define RAY_T1 3.3f
#define DT ((RAY_T1-RAY_T0)/(float)(SAMP-1))

#define LOG2E 1.44269504088896340736f
#define LN2   0.69314718055994530942f

typedef __attribute__((ext_vector_type(16))) _Float16 v16h;
typedef __attribute__((ext_vector_type(8)))  float    v8f;

__device__ __forceinline__ int imin(int a, int b) { return a < b ? a : b; }
__device__ __forceinline__ int imax(int a, int b) { return a > b ? a : b; }

// Raw hardware transcendentals (v_exp_f32 = exp2, v_log_f32 = log2), no
// libm denorm fixups. All arguments here are provably outside denormal range.
__device__ __forceinline__ float fast_exp(float x) {        // e^x
  return __builtin_amdgcn_exp2f(x * LOG2E);
}
__device__ __forceinline__ float softplus_f(float x) {      // ln(1+e^x)
  // stable: max(x,0) + ln(1 + e^-|x|);  log2 arg is in [1,2]
  return fmaxf(x, 0.f) +
         LN2 * __builtin_amdgcn_logf(1.f + fast_exp(-fabsf(x)));
}
__device__ __forceinline__ float sigmoid_f(float x) {
  return __builtin_amdgcn_rcpf(1.f + fast_exp(-x));
}
__device__ __forceinline__ v8f wmma16(v16h a, v16h b, v8f c) {
  // (neg_a, A, neg_b, B, c_mod, C, reuse_a, reuse_b)
  return __builtin_amdgcn_wmma_f32_16x16x32_f16(false, a, false, b, (short)0, c,
                                                false, false);
}

// One MLP pass over 8 rays x 48 samples: tri-plane bilinear gather -> WMMA MLP.
// SLOTBASE=0 : coarse (depth from linspace), SLOTBASE=48 : fine (depth from sh_zf)
template <int SLOTBASE>
__device__ __forceinline__ void mlp_pass(
    const float* __restrict__ planes, const float* __restrict__ ro,
    const float* __restrict__ rd,
    const v16h* w1f, const v16h (*w2f)[2],
    const float* b1v, const float* b2v,
    float* sh_scratch, float* sh_sig, _Float16* sh_rgb, const float* sh_zf,
    int wg, int nb, int wave, int lane)
{
  const int ln = lane & 15, lh = lane >> 4;
  _Float16* hs = ((_Float16*)sh_scratch) + wave * 1024;  // 16x64 f16 per wave

  for (int q = 0; q < TILES_PER_WAVE; ++q) {
    const int tile   = wave * TILES_PER_WAVE + q;
    const int ptbase = tile * 16;
    const int pt = ptbase + ln;
    const int rl = pt / SAMP;
    const int s  = pt - rl * SAMP;
    const int gid = wg * RAYS_PER_WG + rl;

    const float ox = ro[gid*3+0], oy = ro[gid*3+1], oz = ro[gid*3+2];
    const float dx = rd[gid*3+0], dy = rd[gid*3+1], dz = rd[gid*3+2];
    const float td = (SLOTBASE == 0) ? (RAY_T0 + (float)s * DT)
                                     : sh_zf[rl * SAMP + s];
    const float px = ox + td*dx, py = oy + td*dy, pz = oz + td*dz;

    // tri-plane bilinear sample: this lane owns 16 of the 32 channels of its
    // point, laid out exactly as the WMMA f16 A-fragment expects.
    float acc[16];
#pragma unroll
    for (int i = 0; i < 16; ++i) acc[i] = 0.f;

#pragma unroll
    for (int pl = 0; pl < 3; ++pl) {
      // plane0:(x,y)  plane1:(x,z)  plane2:(z,x)   (BOX_WARP scale == 1)
      const float u = (pl == 2) ? pz : px;
      const float v = (pl == 0) ? py : ((pl == 1) ? pz : px);
      const float xi = (u + 1.f) * 128.f - 0.5f;
      const float yi = (v + 1.f) * 128.f - 0.5f;
      const float xf = floorf(xi), yf = floorf(yi);
      const float wx = xi - xf,   wy = yi - yf;
      const int x0 = (int)xf, y0 = (int)yf, x1 = x0 + 1, y1 = y0 + 1;
      const float vx0 = (x0 >= 0 && x0 < HRES) ? 1.f : 0.f;
      const float vx1 = (x1 >= 0 && x1 < HRES) ? 1.f : 0.f;
      const float vy0 = (y0 >= 0 && y0 < HRES) ? 1.f : 0.f;
      const float vy1 = (y1 >= 0 && y1 < HRES) ? 1.f : 0.f;
      const int cx0 = imin(imax(x0,0),HRES-1), cx1 = imin(imax(x1,0),HRES-1);
      const int cy0 = imin(imax(y0,0),HRES-1), cy1 = imin(imax(y1,0),HRES-1);
      const float w00 = (1.f-wx)*(1.f-wy)*vx0*vy0;
      const float w10 = wx*(1.f-wy)*vx1*vy0;
      const float w01 = (1.f-wx)*wy*vx0*vy1;
      const float w11 = wx*wy*vx1*vy1;
      // 32-bit per-lane offsets on a wave-uniform (scalar) plane base
      const int o00 = cy0*HRES + cx0, o10 = cy0*HRES + cx1;
      const int o01 = cy1*HRES + cx0, o11 = cy1*HRES + cx1;
      const float* bp = planes + (long)(nb*3 + pl) * CCH * (HRES*(long)HRES);
#pragma unroll
      for (int i = 0; i < 16; ++i) {
        const int ch = (i >> 3)*16 + lh*8 + (i & 7);   // A-fragment channel map
        const int cb = ch * (HRES*HRES);
        acc[i] += w00*bp[cb + o00] + w10*bp[cb + o10] +
                  w01*bp[cb + o01] + w11*bp[cb + o11];
      }
    }
    v16h afrag;
#pragma unroll
    for (int i = 0; i < 16; ++i) afrag[i] = (_Float16)(acc[i] * (1.f/3.f));

    const v8f zero = {0.f,0.f,0.f,0.f,0.f,0.f,0.f,0.f};
    // GEMM1: X(16x32) @ W1(32x64)
    v8f hacc[4];
#pragma unroll
    for (int nt = 0; nt < 4; ++nt) hacc[nt] = wmma16(afrag, w1f[nt], zero);

    // bias + softplus -> stage H (16x64) in wave-private LDS, f16
#pragma unroll
    for (int nt = 0; nt < 4; ++nt) {
      const int col = nt*16 + ln;
#pragma unroll
      for (int r = 0; r < 8; ++r) {
        const int m = r + 8*lh;   // D-matrix row layout
        hs[m*64 + col] = (_Float16)softplus_f(hacc[nt][r] + b1v[nt]);
      }
    }
    // re-load H as A-fragments (M-striped), K = 64 split in two 32-chunks
    v16h a2[2];
#pragma unroll
    for (int kc = 0; kc < 2; ++kc) {
#pragma unroll
      for (int i = 0; i < 16; ++i) {
        const int k = kc*32 + (i >> 3)*16 + lh*8 + (i & 7);
        a2[kc][i] = hs[ln*64 + k];
      }
    }
    // GEMM2: H(16x64) @ W2(64x48, cols>=33 zero-padded)
#pragma unroll
    for (int nt = 0; nt < 3; ++nt) {
      v8f o = wmma16(a2[0], w2f[nt][0], zero);
      o     = wmma16(a2[1], w2f[nt][1], o);
      const int col = nt*16 + ln;
#pragma unroll
      for (int r = 0; r < 8; ++r) {
        const int m  = r + 8*lh;
        const int p2 = ptbase + m;
        const int rl2 = p2 / SAMP;
        const int s2  = p2 - rl2*SAMP;
        const int slot = rl2*96 + SLOTBASE + s2;
        const float val = o[r] + b2v[nt];
        if (col == 0)
          sh_sig[slot] = val;
        else if (col < 33)
          sh_rgb[slot*CCH + (col-1)] =
              (_Float16)(sigmoid_f(val) * 1.002f - 0.001f);
      }
    }
  }
}

__global__ __launch_bounds__(BLOCK)
void triplane_importance_renderer(const float* __restrict__ planes,
                                  const float* __restrict__ ro,
                                  const float* __restrict__ rd,
                                  const float* __restrict__ w1,
                                  const float* __restrict__ b1,
                                  const float* __restrict__ w2,
                                  const float* __restrict__ b2,
                                  float* __restrict__ out)
{
  __shared__ float          sh_scratch[2048];                 // 8 KB: H-stage / pdf scratch
  __shared__ float          sh_sig[RAYS_PER_WG * 96];         // 3 KB
  __shared__ float          sh_zf [RAYS_PER_WG * SAMP];       // 1.5 KB
  __shared__ unsigned char  sh_ord[RAYS_PER_WG * 96];         // 0.75 KB
  __shared__ _Float16       sh_rgb[RAYS_PER_WG * 96 * CCH];   // 48 KB

  const int t    = threadIdx.x;
  const int wave = t >> 5, lane = t & 31;
  const int ln   = lane & 15, lh = lane >> 4;
  const int wg   = blockIdx.x;
  const int nb   = wg >> 9;   // scalar batch index: 512 workgroups per batch

  // ---- preload weight B-fragments (f16) and biases into registers ----
  v16h w1f[4];            // W1: 32x64 -> 4 N-tiles
  v16h w2f[3][2];         // W2: 64x48 -> [ntile][kchunk]
  float b1v[4], b2v[3];
#pragma unroll
  for (int nt = 0; nt < 4; ++nt) {
#pragma unroll
    for (int i = 0; i < 16; ++i) {
      const int k = (i >> 3)*16 + lh*8 + (i & 7); // B-frag: K = (i/8)*16 + (lane/16)*8 + i%8
      w1f[nt][i] = (_Float16)w1[k*64 + nt*16 + ln];
    }
    b1v[nt] = b1[nt*16 + ln];
  }
#pragma unroll
  for (int nt = 0; nt < 3; ++nt) {
    const int col = nt*16 + ln;
#pragma unroll
    for (int kc = 0; kc < 2; ++kc) {
#pragma unroll
      for (int i = 0; i < 16; ++i) {
        const int k = kc*32 + (i >> 3)*16 + lh*8 + (i & 7);
        w2f[nt][kc][i] = (col < 33) ? (_Float16)w2[k*33 + col]
                                    : (_Float16)0.f;
      }
    }
    b2v[nt] = (col < 33) ? b2[col] : 0.f;
  }

  // ---- coarse pass ----
  mlp_pass<0>(planes, ro, rd, w1f, w2f, b1v, b2v,
              sh_scratch, sh_sig, sh_rgb, sh_zf, wg, nb, wave, lane);
  __syncthreads();

  // ---- ray-march weights + PDF importance sampling (1 thread / ray) ----
  if (t < RAYS_PER_WG) {
    float* w   = sh_scratch + t*128;   // 47 weights
    float* cdf = w + 47;               // 46 cdf entries
    float T = 1.f;
#pragma unroll 1
    for (int i = 0; i < NSEG; ++i) {
      const float sm    = 0.5f*(sh_sig[t*96 + i] + sh_sig[t*96 + i + 1]);
      const float dens  = softplus_f(sm - 1.f);
      const float alpha = 1.f - fast_exp(-dens*DT);
      w[i] = alpha*T;
      T *= (1.f - alpha + 1e-10f);
    }
    cdf[0] = 0.f;
    float run = 0.f;
#pragma unroll 1
    for (int i = 1; i <= 45; ++i) {          // sw[1..45] only (sw[:,1:-1])
      const float m0 = fmaxf(w[i-1], w[i]);
      const float m1 = fmaxf(w[i],   w[i+1]);
      run += 0.5f*(m0 + m1) + 0.01f + 1e-5f;  // + sample_pdf eps
      cdf[i] = run;
    }
    const float inv = 1.f / run;
    int ptr = 1;
#pragma unroll 1
    for (int j = 0; j < SAMP; ++j) {
      const float u = (float)j * (1.f/47.f);
      while (ptr < 46 && cdf[ptr]*inv <= u) ++ptr;   // searchsorted 'right'
      const int below = ptr - 1;
      const int above = imin(ptr, 45);
      const float cb = cdf[below]*inv, ca = cdf[above]*inv;
      const float zb = RAY_T0 + ((float)below + 0.5f)*DT;  // z_mid bins
      const float za = RAY_T0 + ((float)above + 0.5f)*DT;
      float denom = ca - cb;
      if (denom < 1e-5f) denom = 1.f;
      sh_zf[t*SAMP + j] = zb + (u - cb)/denom*(za - zb);
    }
  }
  __syncthreads();

  // ---- fine pass ----
  mlp_pass<SAMP>(planes, ro, rd, w1f, w2f, b1v, b2v,
                 sh_scratch, sh_sig, sh_rgb, sh_zf, wg, nb, wave, lane);
  __syncthreads();

  // ---- stable merge of sorted coarse + sorted fine depths ----
  if (t < RAYS_PER_WG) {
    int i = 0, j = 0;
#pragma unroll 1
    for (int k = 0; k < 96; ++k) {
      const float zc = (i < SAMP) ? (RAY_T0 + (float)i*DT) : 3.0e38f;
      const float zf = (j < SAMP) ? sh_zf[t*SAMP + j]      : 3.0e38f;
      if (zc <= zf) { sh_ord[t*96 + k] = (unsigned char)i;          ++i; }
      else          { sh_ord[t*96 + k] = (unsigned char)(SAMP + j); ++j; }
    }
  }
  __syncthreads();

  // ---- final ray march: 16 lanes per ray, 2 channels per lane ----
  {
    const int r  = t >> 4;
    const int lg = t & 15;
    const int c0i = 2*lg, c1i = 2*lg + 1;
    float T = 1.f, cw = 0.f, cd = 0.f, a0 = 0.f, a1 = 0.f;
    int   i0 = sh_ord[r*96];
    float d0 = (i0 < SAMP) ? (RAY_T0 + (float)i0*DT) : sh_zf[r*SAMP + (i0-SAMP)];
    float s0 = sh_sig[r*96 + i0];
    float r00 = (float)sh_rgb[(r*96 + i0)*CCH + c0i];
    float r01 = (float)sh_rgb[(r*96 + i0)*CCH + c1i];
#pragma unroll 1
    for (int k = 1; k < 96; ++k) {
      const int   i1 = sh_ord[r*96 + k];
      const float d1 = (i1 < SAMP) ? (RAY_T0 + (float)i1*DT)
                                   : sh_zf[r*SAMP + (i1-SAMP)];
      const float s1  = sh_sig[r*96 + i1];
      const float r10 = (float)sh_rgb[(r*96 + i1)*CCH + c0i];
      const float r11 = (float)sh_rgb[(r*96 + i1)*CCH + c1i];
      const float delta = d1 - d0;
      const float dens  = softplus_f(0.5f*(s0 + s1) - 1.f);
      const float alpha = 1.f - fast_exp(-dens*delta);
      const float wgt   = alpha*T;
      T *= (1.f - alpha + 1e-10f);
      cw += wgt;
      cd += wgt*0.5f*(d0 + d1);
      a0 += wgt*0.5f*(r00 + r10);
      a1 += wgt*0.5f*(r01 + r11);
      d0 = d1; s0 = s1; r00 = r10; r01 = r11;
    }
    const int gid = wg*RAYS_PER_WG + r;
    out[gid*CCH + c0i] = a0*2.f - 1.f;
    out[gid*CCH + c1i] = a1*2.f - 1.f;
    if (lg == 0) {
      float dd = cd / cw;
      if (dd != dd) dd = 1e30f;                       // nan -> +inf
      dd = fminf(fmaxf(dd, RAY_T0), RAY_T1);          // clip to depth range
      out[TOTAL_RAYS*CCH + gid]              = dd;    // depth
      out[TOTAL_RAYS*CCH + TOTAL_RAYS + gid] = cw;    // weight total
    }
  }
}

extern "C" void kernel_launch(void* const* d_in, const int* in_sizes, int n_in,
                              void* d_out, int out_size, void* d_ws, size_t ws_size,
                              hipStream_t stream) {
  (void)in_sizes; (void)n_in; (void)out_size; (void)d_ws; (void)ws_size;
  const float* planes = (const float*)d_in[0];
  const float* ro     = (const float*)d_in[1];
  const float* rd     = (const float*)d_in[2];
  const float* w1     = (const float*)d_in[3];
  const float* b1     = (const float*)d_in[4];
  const float* w2     = (const float*)d_in[5];
  const float* b2     = (const float*)d_in[6];
  float* out = (float*)d_out;

  dim3 grid(TOTAL_RAYS / RAYS_PER_WG);   // 2048 workgroups
  dim3 block(BLOCK);                     // 128 threads = 4 wave32
  triplane_importance_renderer<<<grid, block, 0, stream>>>(
      planes, ro, rd, w1, b1, w2, b2, out);
}
